// NeuronLevelModels_49727131353412
// MI455X (gfx1250) — compile-verified, matrix operand order checked by reading, not run on hardware
//
#include <hip/hip_runtime.h>
#include <stdint.h>

// MI455X / gfx1250, wave32. Memory-bound per-neuron MLP:
//   h = gelu(x[b,n,:] @ w1[n,:,:]);  out[b,n] = h @ w2[n,:,0]
// Layer 1 via v_wmma_f32_16x16x32_bf16 (4 per wave per tile);
// GELU branch-free on v_exp_f32 + v_rcp_f32; layer 2 VALU + shuffle reduce.

typedef __attribute__((ext_vector_type(16))) __bf16 v16bf;
typedef __attribute__((ext_vector_type(8)))  float  v8f;

#define NEURONS 4096
#define HIST    64
#define MID     128
#define BATCH   1024
#define BSPLIT  4                        // batch quarters across gridDim.y
#define ROWS    32                       // batch rows per tile (2 WMMA row tiles)
#define ITERS   (BATCH / BSPLIT / ROWS)  // 8 tiles per block

union Frag {
    v16bf    v;       // 16 bf16 = 32 bytes = 8 VGPRs
    uint16_t e[16];
    uint4    q[2];
};

static __device__ inline uint16_t f2bf(float f) {
    // round-to-nearest-even fp32 -> bf16
    uint32_t u = __float_as_uint(f);
    u += 0x7FFFu + ((u >> 16) & 1u);
    return (uint16_t)(u >> 16);
}

static __device__ inline uint32_t pk_bf16(float a, float b) {
#if __has_builtin(__builtin_amdgcn_cvt_pk_bf16_f32)
    auto r = __builtin_amdgcn_cvt_pk_bf16_f32(a, b);   // v_cvt_pk_bf16_f32
    uint32_t u; __builtin_memcpy(&u, &r, 4);
    return u;
#else
    return (uint32_t)f2bf(a) | ((uint32_t)f2bf(b) << 16);
#endif
}

// Branch-free GELU (tanh form) on hardware transcendentals:
//   gelu(x) = x * rcp(1 + exp2(-2.3022082*(x + 0.044715 x^3)))
// exp2(-inf)=0 -> g=x ; exp2(+inf)=inf -> rcp=0 -> g=0 : saturates correctly.
static __device__ inline float gelu_fast(float x) {
    const float t     = x * x;
    const float inner = x * __builtin_fmaf(0.044715f, t, 1.0f);
    const float e     = __builtin_amdgcn_exp2f(-2.302208198f * inner);
    return x * __builtin_amdgcn_rcpf(1.0f + e);
}

__global__ __launch_bounds__(256)
void neuron_mlp_kernel(const float* __restrict__ x,
                       const float* __restrict__ w1,
                       const float* __restrict__ w2,
                       float* __restrict__ out) {
    const int n    = blockIdx.x;           // neuron
    const int tid  = threadIdx.x;          // 0..255 (8 waves)
    const int wave = tid >> 5;             // 0..7 -> M-column tile of 16
    const int lane = tid & 31;             // wave32 lane
    const int half = lane >> 4;            // 0: lanes 0-15, 1: lanes 16-31
    const int col  = lane & 15;            // column within 16 / row within 16
    const int wcol0 = wave * 16;           // this wave's MID column base

    // ---- Per-wave persistent B fragments of w1 (K=0..31 and K=32..63) ----
    // ISA 16-bit matrix layout: element e -> K = (e/8)*16 + half*8 + (e%8)
    const float* w1p = w1 + (size_t)n * HIST * MID;
    Frag b0, b1;
#pragma unroll
    for (int e = 0; e < 16; ++e) {
        const int k = ((e >> 3) << 4) + (half << 3) + (e & 7);
        b0.e[e] = f2bf(w1p[(size_t)k        * MID + wcol0 + col]);
        b1.e[e] = f2bf(w1p[(size_t)(k + 32) * MID + wcol0 + col]);
    }
    const float w2v = w2[(size_t)n * MID + wcol0 + col];

    __shared__ __align__(16) uint16_t xs[ROWS * HIST];  // bf16 x tile, 4 KB
    __shared__ float sum_lds[ROWS];

    const int ldrow = tid >> 4;        // 0..15 (also handles ldrow+16)
    const int ldc4  = (tid & 15) * 4;  // 0..60: H offset (float4 chunk)

    const uint4* xsq = reinterpret_cast<const uint4*>(xs);
    // uint4-granule index into xs: (row*64 + kbase + g*16 + half*8)/8
    const int abase = (lane & 15) * (HIST / 8) + half;   // row tile 0
    const int abase2 = abase + 16 * (HIST / 8);          // row tile 1 (rows 16-31)

    for (int it = 0; it < ITERS; ++it) {
        const int brow0 = blockIdx.y * (BATCH / BSPLIT) + it * ROWS;

        __syncthreads();                       // prev out-write done
        if (tid < ROWS) sum_lds[tid] = 0.0f;

        // ---- cooperative x tile load: 32 rows x 64 H, fp32 -> bf16 in LDS ----
        const float* xp0 = x + ((size_t)(brow0 + ldrow) * NEURONS + n) * HIST + ldc4;
        const float* xp1 = xp0 + (size_t)16 * NEURONS * HIST;
        const float4 v0 = *reinterpret_cast<const float4*>(xp0);
        const float4 v1 = *reinterpret_cast<const float4*>(xp1);
        *reinterpret_cast<uint2*>(&xs[ldrow * HIST + ldc4]) =
            make_uint2(pk_bf16(v0.x, v0.y), pk_bf16(v0.z, v0.w));
        *reinterpret_cast<uint2*>(&xs[(ldrow + 16) * HIST + ldc4]) =
            make_uint2(pk_bf16(v1.x, v1.y), pk_bf16(v1.z, v1.w));

        if (it + 1 < ITERS) {                  // overlap next tile fetch
            __builtin_prefetch(xp0 + (size_t)ROWS * NEURONS * HIST, 0, 3);
            __builtin_prefetch(xp1 + (size_t)ROWS * NEURONS * HIST, 0, 3);
        }
        __syncthreads();

        // ---- layer 1: two 16x16 f32 row tiles, K=64 -> 2 WMMAs each ----
        Frag a0, a1;
        a0.q[0] = xsq[abase + 0];
        a0.q[1] = xsq[abase + 2];
        a1.q[0] = xsq[abase + 4];
        a1.q[1] = xsq[abase + 6];
        v8f c0 = {};
        c0 = __builtin_amdgcn_wmma_f32_16x16x32_bf16(false, a0.v, false, b0.v,
                                                     (short)0, c0, false, false);
        c0 = __builtin_amdgcn_wmma_f32_16x16x32_bf16(false, a1.v, false, b1.v,
                                                     (short)0, c0, false, false);

        Frag a2, a3;
        a2.q[0] = xsq[abase2 + 0];
        a2.q[1] = xsq[abase2 + 2];
        a3.q[0] = xsq[abase2 + 4];
        a3.q[1] = xsq[abase2 + 6];
        v8f c1 = {};
        c1 = __builtin_amdgcn_wmma_f32_16x16x32_bf16(false, a2.v, false, b0.v,
                                                     (short)0, c1, false, false);
        c1 = __builtin_amdgcn_wmma_f32_16x16x32_bf16(false, a3.v, false, b1.v,
                                                     (short)0, c1, false, false);

        // ---- GELU + layer 2 partial products (branch-free) ----
        float partial[16];
#pragma unroll
        for (int r = 0; r < 8; ++r) {
            partial[r]     = gelu_fast(c0[r]) * w2v;
            partial[8 + r] = gelu_fast(c1[r]) * w2v;
        }

        // butterfly reduce over the 16 lanes of each half
#pragma unroll
        for (int m = 1; m <= 8; m <<= 1) {
#pragma unroll
            for (int r = 0; r < 16; ++r)
                partial[r] += __shfl_xor(partial[r], m, 32);
        }
        if ((lane & 15) == 0) {
            const int rb = half * 8;
#pragma unroll
            for (int r = 0; r < 8; ++r) {
                atomicAdd(&sum_lds[rb + r],      partial[r]);       // rows 0-15
                atomicAdd(&sum_lds[16 + rb + r], partial[8 + r]);   // rows 16-31
            }
        }
        __syncthreads();

        if (tid < ROWS)
            out[(size_t)(brow0 + tid) * NEURONS + n] = sum_lds[tid];
    }
}

extern "C" void kernel_launch(void* const* d_in, const int* in_sizes, int n_in,
                              void* d_out, int out_size, void* d_ws, size_t ws_size,
                              hipStream_t stream) {
    (void)in_sizes; (void)n_in; (void)out_size; (void)d_ws; (void)ws_size;
    const float* x  = (const float*)d_in[0];
    const float* w1 = (const float*)d_in[1];
    const float* w2 = (const float*)d_in[2];
    float* out = (float*)d_out;

    dim3 grid(NEURONS, BSPLIT);
    dim3 block(256);
    neuron_mlp_kernel<<<grid, block, 0, stream>>>(x, w1, w2, out);
}
